// GlobalVariableLRSpatioTemporalGNN_40037685133367
// MI455X (gfx1250) — matrix operand order repaired and checked by reading, not compiled, not run on hardware
//
#include <hip/hip_runtime.h>
#include <hip/hip_bf16.h>
#include <stdint.h>

// ---------------------------------------------------------------------------
// Spatio-temporal GNN forward for MI455X (gfx1250, wave32, WMMA).
// All dense math runs through v_wmma_f32_16x16x32_bf16 (bf16 in, f32 accum).
// Input leaf order assumes jax.tree_util.tree_leaves(setup_inputs()) i.e.
// dict keys sorted alphabetically at every level (141 leaves total).
// ---------------------------------------------------------------------------

typedef __bf16 v16bf __attribute__((ext_vector_type(16)));
typedef float  v8f   __attribute__((ext_vector_type(8)));

#define DEV __device__ __forceinline__

DEV unsigned short f2bfu(float f) {          // f32 -> bf16 bits, RNE
  unsigned u = __builtin_bit_cast(unsigned, f);
  u += 0x7fffu + ((u >> 16) & 1u);
  return (unsigned short)(u >> 16);
}
DEV float bfu2f(unsigned short h) {
  unsigned u = ((unsigned)h) << 16;
  return __builtin_bit_cast(float, u);
}
DEV __bf16 u2bf(unsigned short h) { return __builtin_bit_cast(__bf16, h); }
DEV v8f vzero8() {
  v8f v;
#pragma unroll
  for (int i = 0; i < 8; ++i) v[i] = 0.f;
  return v;
}

// constants of the model
#define GN   6        // B*T graphs
#define NND  1024     // nodes
#define NE   9216     // edges (N*DEG + N)
#define HH   384
#define MNODE (GN*NND)   // 6144
#define MEDGE (GN*NE)    // 55296

// ---------------------------------------------------------------------------
// Generic WMMA GEMM: C[M,N] = act(A[M,K] @ W[K,N] + bias)
// amode 0: A = plain ptr (f32 or bf16 via abf16), row stride lda
// amode 1: A cols [0,384) from Aptr (stride 384), [384,768) from A1 (stride 384)
// amode 2: edge concat: row m=(g*NE+el): [h[src],h[tgt],ea] (Aptr=h f32, A1=ea)
// Output: f32 to Cf (if non-null) else bf16 bits to Cb. act: 0 none,1 relu,2 sigmoid
// ---------------------------------------------------------------------------
__global__ __launch_bounds__(256) void k_gemm(
    const void* __restrict__ Aptr, const float* __restrict__ A1,
    const int* __restrict__ esrc, const int* __restrict__ etgt,
    const float* __restrict__ W, const float* __restrict__ bias,
    float* __restrict__ Cf, unsigned short* __restrict__ Cb,
    int amode, int abf16, int lda, int ldc,
    int M, int Nn, int K, int act)
{
  __shared__ unsigned short As[64][34];   // [m][k] bf16 bits
  __shared__ unsigned short Bs[64][34];   // [n][k] bf16 bits (W transposed tile)
  const int tid  = threadIdx.x;
  const int lane = tid & 31;
  const int wave = tid >> 5;
  const int wm = wave & 3;       // 4 waves along M (16 rows each)
  const int wn = wave >> 2;      // 2 waves along N (32 cols each)
  const int m0 = blockIdx.y * 64, n0 = blockIdx.x * 64;

  v8f acc0 = vzero8(), acc1 = vzero8();

  for (int k0 = 0; k0 < K; k0 += 32) {
    for (int i = tid; i < 64 * 32; i += 256) {
      int r = i >> 5, kk = i & 31;
      int gm = m0 + r, gk = k0 + kk;
      float av = 0.f;
      if (gm < M && gk < K) {
        if (amode == 0) {
          av = abf16 ? bfu2f(((const unsigned short*)Aptr)[(long)gm * lda + gk])
                     : ((const float*)Aptr)[(long)gm * lda + gk];
        } else if (amode == 1) {
          av = (gk < 384) ? ((const float*)Aptr)[(long)gm * 384 + gk]
                          : A1[(long)gm * 384 + (gk - 384)];
        } else {
          int g = gm / NE, el = gm % NE;
          if (gk < 384)      av = ((const float*)Aptr)[((long)g * NND + esrc[el]) * HH + gk];
          else if (gk < 768) av = ((const float*)Aptr)[((long)g * NND + etgt[el]) * HH + (gk - 384)];
          else               av = A1[(long)el * 3 + (gk - 768)];
        }
      }
      As[r][kk] = f2bfu(av);
      int gn = n0 + r;
      float bv = 0.f;
      if (gn < Nn && gk < K) bv = W[(long)gk * Nn + gn];
      Bs[r][kk] = f2bfu(bv);
    }
    __syncthreads();
    // fragments per ISA layout (CDNA5 ISA 7.12.2)
    v16bf af, bf0, bf1;
    const int arow = wm * 16 + (lane & 15);
    const int kb  = (lane >= 16) ? 8  : 0;  // A K-offset
    const int kb2 = (lane >= 16) ? 16 : 0;  // unused; B also uses kb
    const int bc0 = wn * 32 + (lane & 15);
#pragma unroll
    for (int e = 0; e < 16; ++e) {
      int kk = (e & 7) + ((e & 8) << 1) + kb;
      af[e]  = u2bf(As[arow][kk]);
      bf0[e] = u2bf(Bs[bc0][kk]);
      bf1[e] = u2bf(Bs[bc0 + 16][kk]);
    }
    acc0 = __builtin_amdgcn_wmma_f32_16x16x32_bf16(false, af, false, bf0, (short)0, acc0, false, false);
    acc1 = __builtin_amdgcn_wmma_f32_16x16x32_bf16(false, af, false, bf1, (short)0, acc1, false, false);
    __syncthreads();
  }
  const int hi8 = (lane >= 16) ? 8 : 0;
#pragma unroll
  for (int r = 0; r < 8; ++r) {
    int gm = m0 + wm * 16 + r + hi8;
    if (gm >= M) continue;
    int gn0 = n0 + wn * 32 + (lane & 15);
    int gn1 = gn0 + 16;
    float v0 = acc0[r], v1 = acc1[r];
    if (bias) {
      if (gn0 < Nn) v0 += bias[gn0];
      if (gn1 < Nn) v1 += bias[gn1];
    }
    if (act == 1) { v0 = fmaxf(v0, 0.f); v1 = fmaxf(v1, 0.f); }
    else if (act == 2) { v0 = 1.f / (1.f + __expf(-v0)); v1 = 1.f / (1.f + __expf(-v1)); }
    if (Cf) {
      if (gn0 < Nn) Cf[(long)gm * ldc + gn0] = v0;
      if (gn1 < Nn) Cf[(long)gm * ldc + gn1] = v1;
    } else {
      if (gn0 < Nn) Cb[(long)gm * ldc + gn0] = f2bfu(v0);
      if (gn1 < Nn) Cb[(long)gm * ldc + gn1] = f2bfu(v1);
    }
  }
}

// ---------------------------------------------------------------------------
// Flash attention: one wave per (graph, head, 16-query tile); 32 keys/chunk.
// S = Q K^T and O += P V both via WMMA; online softmax; optional edge bitmask.
// Q/K/V/O are f32 [G*N, 384], head channels contiguous at head*HD.
// ---------------------------------------------------------------------------
template<int HD, int HDP>
__global__ __launch_bounds__(32) void k_flash(
    const float* __restrict__ Q, const float* __restrict__ K,
    const float* __restrict__ V, float* __restrict__ O,
    float scale, const unsigned int* __restrict__ maskw)
{
  constexpr int NKC = HDP / 32;   // K-chunks along head dim for S
  constexpr int NCT = HD / 16;    // 16-wide output col tiles
  __shared__ unsigned short Qs[16][HDP];
  __shared__ unsigned short KVs[32][HDP];
  __shared__ unsigned short Ps[16][32];
  const int lane = threadIdx.x;
  const int q0   = blockIdx.x * 16;
  const int head = blockIdx.y;
  const int g    = blockIdx.z;
  const long rowbase = (long)g * NND * HH;
  const int hoff = head * HD;
  const int l15 = lane & 15;
  const int kb  = (lane >= 16) ? 8  : 0;
  const int kb2 = (lane >= 16) ? 16 : 0;
  const int hi8 = (lane >= 16) ? 8  : 0;

  for (int i = lane; i < 16 * HDP; i += 32) {
    int r = i / HDP, c = i % HDP;
    Qs[r][c] = f2bfu((c < HD) ? Q[rowbase + (long)(q0 + r) * HH + hoff + c] : 0.f);
  }
  __syncthreads();
  v16bf qf[NKC];
#pragma unroll
  for (int c = 0; c < NKC; ++c)
#pragma unroll
    for (int e = 0; e < 16; ++e)
      qf[c][e] = u2bf(Qs[l15][c * 32 + (e & 7) + ((e & 8) << 1) + kb]);

  float mrow[8], lrow[8];
#pragma unroll
  for (int r = 0; r < 8; ++r) { mrow[r] = -1e30f; lrow[r] = 0.f; }
  v8f oacc[NCT];
#pragma unroll
  for (int c = 0; c < NCT; ++c) oacc[c] = vzero8();

  for (int j0 = 0; j0 < NND; j0 += 32) {
    __syncthreads();
    for (int i = lane; i < 32 * HDP; i += 32) {     // K chunk -> LDS
      int r = i / HDP, c = i % HDP;
      KVs[r][c] = f2bfu((c < HD) ? K[rowbase + (long)(j0 + r) * HH + hoff + c] : 0.f);
    }
    __syncthreads();
    v8f s0 = vzero8(), s1 = vzero8();
#pragma unroll
    for (int c = 0; c < NKC; ++c) {
      v16bf kf0, kf1;
#pragma unroll
      for (int e = 0; e < 16; ++e) {
        kf0[e] = u2bf(KVs[l15][c * 32 + (e & 7) + ((e & 8) << 1) + kb]);        // keys j0..j0+15
        kf1[e] = u2bf(KVs[16 + l15][c * 32 + (e & 7) + ((e & 8) << 1) + kb]);   // keys j0+16..j0+31
      }
      s0 = __builtin_amdgcn_wmma_f32_16x16x32_bf16(false, qf[c], false, kf0, (short)0, s0, false, false);
      s1 = __builtin_amdgcn_wmma_f32_16x16x32_bf16(false, qf[c], false, kf1, (short)0, s1, false, false);
    }
    float sv0[8], sv1[8], mx[8];
#pragma unroll
    for (int r = 0; r < 8; ++r) {
      float a = s0[r] * scale, b = s1[r] * scale;
      if (maskw) {
        int q = q0 + r + hi8;
        int k0i = j0 + l15, k1i = k0i + 16;
        if (!((maskw[q * 32 + (k0i >> 5)] >> (k0i & 31)) & 1u)) a = -1e30f;
        if (!((maskw[q * 32 + (k1i >> 5)] >> (k1i & 31)) & 1u)) b = -1e30f;
      }
      sv0[r] = a; sv1[r] = b;
      float m = fmaxf(a, b);
      for (int off = 8; off > 0; off >>= 1) m = fmaxf(m, __shfl_xor(m, off, 32));
      mx[r] = m;
    }
#pragma unroll
    for (int r = 0; r < 8; ++r) {
      float mnew  = fmaxf(mrow[r], mx[r]);
      float alpha = __expf(mrow[r] - mnew);
      float p0 = (sv0[r] <= -5e29f) ? 0.f : __expf(sv0[r] - mnew);
      float p1 = (sv1[r] <= -5e29f) ? 0.f : __expf(sv1[r] - mnew);
      float rs = p0 + p1;
      for (int off = 8; off > 0; off >>= 1) rs += __shfl_xor(rs, off, 32);
      lrow[r] = lrow[r] * alpha + rs;
      mrow[r] = mnew;
#pragma unroll
      for (int c = 0; c < NCT; ++c) oacc[c][r] *= alpha;
      Ps[r + hi8][l15]      = f2bfu(p0);
      Ps[r + hi8][16 + l15] = f2bfu(p1);
    }
    __syncthreads();
    for (int i = lane; i < 32 * HDP; i += 32) {     // V chunk -> LDS (reuse)
      int r = i / HDP, c = i % HDP;
      KVs[r][c] = f2bfu((c < HD) ? V[rowbase + (long)(j0 + r) * HH + hoff + c] : 0.f);
    }
    __syncthreads();
    v16bf pf;
#pragma unroll
    for (int e = 0; e < 16; ++e)
      pf[e] = u2bf(Ps[l15][(e & 7) + ((e & 8) << 1) + kb]);
#pragma unroll
    for (int c = 0; c < NCT; ++c) {
      v16bf vf;
#pragma unroll
      for (int e = 0; e < 16; ++e)
        vf[e] = u2bf(KVs[(e & 7) + ((e & 8) << 1) + kb][c * 16 + l15]);
      oacc[c] = __builtin_amdgcn_wmma_f32_16x16x32_bf16(false, pf, false, vf, (short)0, oacc[c], false, false);
    }
  }
#pragma unroll
  for (int c = 0; c < NCT; ++c)
#pragma unroll
    for (int r = 0; r < 8; ++r)
      O[rowbase + (long)(q0 + r + hi8) * HH + hoff + c * 16 + l15] = oacc[c][r] / lrow[r];
}

// ---------------------------------------------------------------------------
// LayerNorm (D=384): mode 0: x = a+b ; mode 1: x = a*b + (1-a)*c (gate combine)
// clips x to [-50,50] first (as the reference does).
// ---------------------------------------------------------------------------
__global__ __launch_bounds__(128) void k_ln(
    const float* __restrict__ a, const float* __restrict__ b, const float* __restrict__ c,
    const float* __restrict__ gw, const float* __restrict__ bw,
    float* __restrict__ out, int mode)
{
  __shared__ float red[128];
  const int tid = threadIdx.x;
  const long base = (long)blockIdx.x * 384;
  float x[3];
#pragma unroll
  for (int i = 0; i < 3; ++i) {
    int col = tid + i * 128;
    float v;
    if (mode == 0) v = a[base + col] + b[base + col];
    else { float gt = a[base + col]; v = gt * b[base + col] + (1.f - gt) * c[base + col]; }
    x[i] = fminf(fmaxf(v, -50.f), 50.f);
  }
  red[tid] = x[0] + x[1] + x[2]; __syncthreads();
  for (int o = 64; o > 0; o >>= 1) { if (tid < o) red[tid] += red[tid + o]; __syncthreads(); }
  float mu = red[0] / 384.f; __syncthreads();
  float d0 = x[0] - mu, d1 = x[1] - mu, d2 = x[2] - mu;
  red[tid] = d0 * d0 + d1 * d1 + d2 * d2; __syncthreads();
  for (int o = 64; o > 0; o >>= 1) { if (tid < o) red[tid] += red[tid + o]; __syncthreads(); }
  float rstd = rsqrtf(red[0] / 384.f + 1e-5f);
#pragma unroll
  for (int i = 0; i < 3; ++i) {
    int col = tid + i * 128;
    out[base + col] = (x[i] - mu) * rstd * gw[col] + bw[col];
  }
}

// --------------------------- small utility kernels -------------------------
__global__ void k_zero_u32(unsigned int* p, int n) {
  int i = blockIdx.x * blockDim.x + threadIdx.x; if (i < n) p[i] = 0u;
}
__global__ void k_count(const int* __restrict__ tgt, int* counts, int e) {
  int i = blockIdx.x * blockDim.x + threadIdx.x; if (i < e) atomicAdd(&counts[tgt[i]], 1);
}
__global__ void k_scan(const int* __restrict__ counts, int* rowstart) {
  if (blockIdx.x == 0 && threadIdx.x == 0) {
    int acc = 0;
    for (int n = 0; n < NND; ++n) { rowstart[n] = acc; acc += counts[n]; }
    rowstart[NND] = acc;
  }
}
__global__ void k_rank(const int* __restrict__ tgt, const int* __restrict__ rowstart,
                       int* order, int e) {
  int i = blockIdx.x * blockDim.x + threadIdx.x; if (i >= e) return;
  int t = tgt[i], r = 0;
  for (int j = 0; j < i; ++j) r += (tgt[j] == t);   // stable rank -> deterministic sum order
  order[rowstart[t] + r] = i;
}
__global__ void k_mask(const int* __restrict__ src, const int* __restrict__ tgt,
                       unsigned int* maskw, int e) {
  int i = blockIdx.x * blockDim.x + threadIdx.x; if (i >= e) return;
  atomicOr(&maskw[src[i] * 32 + (tgt[i] >> 5)], 1u << (tgt[i] & 31));
}
__global__ __launch_bounds__(384) void k_agg(
    const float* __restrict__ msgs, const int* __restrict__ rowstart,
    const int* __restrict__ order, float* __restrict__ agg)
{
  int n = blockIdx.x, g = blockIdx.y, c = threadIdx.x;
  float s = 0.f;
  int b = rowstart[n], en = rowstart[n + 1];
  for (int i = b; i < en; ++i) {
    int e = order[i];
    s += msgs[((long)g * NE + e) * HH + c];
  }
  agg[((long)g * NND + n) * HH + c] = s;
}
__global__ void k_permute_ht(const float* __restrict__ in, float* __restrict__ out) {
  long idx = (long)blockIdx.x * blockDim.x + threadIdx.x;
  if (idx >= (long)MNODE * HH) return;
  long m = idx / HH; int cc = (int)(idx % HH);
  int t = (int)(m / NND), n = (int)(m % NND);
  out[((long)n * 6 + t) * HH + cc] = in[idx];
}
// temporal attention, seq len T=6, 8 heads x 48 — negligible FLOPs, VALU kernel
__global__ __launch_bounds__(256) void k_tattn(
    const float* __restrict__ Q, const float* __restrict__ K,
    const float* __restrict__ V, float* __restrict__ O, float scale)
{
  int idx = blockIdx.x * blockDim.x + threadIdx.x;
  if (idx >= NND * 8 * 6) return;
  int node = idx / 48; int rem = idx % 48; int head = rem / 6; int tq = rem % 6;
  long qrow = ((long)node * 6 + tq) * HH + head * 48;
  float s[6]; float mx = -1e30f;
  for (int tk = 0; tk < 6; ++tk) {
    long krow = ((long)node * 6 + tk) * HH + head * 48;
    float d = 0.f;
    for (int c = 0; c < 48; ++c) d += Q[qrow + c] * K[krow + c];
    s[tk] = d * scale; mx = fmaxf(mx, s[tk]);
  }
  float sum = 0.f;
  for (int tk = 0; tk < 6; ++tk) { s[tk] = __expf(fmaxf(s[tk] - mx, -50.f)); sum += s[tk]; }
  float inv = 1.f / sum;
  for (int c = 0; c < 48; ++c) {
    float acc = 0.f;
    for (int tk = 0; tk < 6; ++tk)
      acc += s[tk] * V[((long)node * 6 + tk) * HH + head * 48 + c];
    O[qrow + c] = acc * inv;
  }
}

// ---------------------------------------------------------------------------
extern "C" void kernel_launch(void* const* d_in, const int* in_sizes, int n_in,
                              void* d_out, int out_size, void* d_ws, size_t ws_size,
                              hipStream_t stream)
{
  (void)in_sizes; (void)out_size; (void)ws_size;
  if (n_in < 141) return;   // leaf-order assumption violated; bail safely

  const float* ea  = (const float*)d_in[0];
  const int*   ei  = (const int*)d_in[1];
  const int* esrc = ei;
  const int* etgt = ei + NE;
  auto F = [&](int i) { return (const float*)d_in[i]; };
  const float* x = F(140);

  // ---- scratch arena ----
  char* base = (char*)d_ws;
  size_t off = 0;
  auto alloc = [&](size_t bytes) -> void* {
    off = (off + 255) & ~(size_t)255;
    void* p = base + off; off += bytes; return p;
  };
  float* msgs = (float*)alloc((size_t)MEDGE * HH * 4);
  unsigned short* hid = (unsigned short*)alloc((size_t)MEDGE * HH * 2);
  float* nb[8];
  for (int i = 0; i < 8; ++i) nb[i] = (float*)alloc((size_t)MNODE * HH * 4);
  unsigned int* maskw = (unsigned int*)alloc((size_t)NND * 32 * 4);
  int* counts   = (int*)alloc(NND * 4);
  int* rowstart = (int*)alloc((NND + 1) * 4);
  int* order    = (int*)alloc(NE * 4);

  const float cscale = 0.10206207261596577f;  // 1/sqrt(96)
  const float sscale = 0.14433756729740643f;  // 1/sqrt(48)

  auto gemm = [&](const void* A, const float* A1, const float* W, const float* bias,
                  float* Cf, unsigned short* Cb, int amode, int abf16,
                  int lda, int ldc, int M, int Nn, int K, int act) {
    dim3 grid((Nn + 63) / 64, (M + 63) / 64);
    k_gemm<<<grid, 256, 0, stream>>>(A, A1, esrc, etgt, W, bias, Cf, Cb,
                                     amode, abf16, lda, ldc, M, Nn, K, act);
  };

  // ---- CSR for deterministic segment-sum + edge mask (h-independent) ----
  k_zero_u32<<<(NND * 32 + 255) / 256, 256, 0, stream>>>(maskw, NND * 32);
  k_zero_u32<<<(NND + 255) / 256, 256, 0, stream>>>((unsigned int*)counts, NND);
  k_count<<<(NE + 255) / 256, 256, 0, stream>>>(etgt, counts, NE);
  k_scan<<<1, 32, 0, stream>>>(counts, rowstart);
  k_rank<<<(NE + 255) / 256, 256, 0, stream>>>(etgt, rowstart, order, NE);
  k_mask<<<(NE + 255) / 256, 256, 0, stream>>>(esrc, etgt, maskw, NE);

  // ---- feature encoder: h = relu(relu(x@fe1)@fe2) ----
  gemm(x, nullptr, F(19), F(18), nb[1], nullptr, 0, 0, 18, 192, MNODE, 192, 18, 1);
  gemm(nb[1], nullptr, F(21), F(20), nb[0], nullptr, 0, 0, 192, 384, MNODE, 384, 192, 1);

  // ---- cluster attention (cp from x[...,11:17]) ----
  gemm(x + 11, nullptr, F(17), F(16), nb[1], nullptr, 0, 0, 18, 384, MNODE, 384, 6, 0);   // cp
  gemm(nb[1], nullptr, F(13), F(12), nb[2], nullptr, 0, 0, 384, 384, MNODE, 384, 384, 0); // Q
  gemm(nb[1], nullptr, F(9),  F(8),  nb[3], nullptr, 0, 0, 384, 384, MNODE, 384, 384, 0); // K
  gemm(nb[1], nullptr, F(15), F(14), nb[4], nullptr, 0, 0, 384, 384, MNODE, 384, 384, 0); // V
  k_flash<96, 96><<<dim3(64, 4, 6), 32, 0, stream>>>(nb[2], nb[3], nb[4], nb[5], cscale, nullptr);
  gemm(nb[5], nullptr, F(11), F(10), nb[6], nullptr, 0, 0, 384, 384, MNODE, 384, 384, 0); // ca
  gemm(nb[0], nb[6], F(3), F(2), nb[1], nullptr, 1, 0, 384, 384, MNODE, 384, 768, 1);     // cl_f1
  gemm(nb[1], nullptr, F(5), F(4), nb[2], nullptr, 0, 0, 384, 384, MNODE, 384, 384, 0);   // cl_f2
  k_ln<<<MNODE, 128, 0, stream>>>(nb[0], nb[2], nullptr, F(7), F(6), nb[0], 0);

  // ---- 8 message-passing layers ----
  for (int l = 0; l < 8; ++l) {
    int p = 24 + 12 * l;  // gate_b,gate_w,ln_b,ln_g,m1_b,m1_w,m2_b,m2_w,u1_b,u1_w,u2_b,u2_w
    gemm(nb[0], ea, F(p + 5), F(p + 4), nullptr, hid, 2, 0, 0, 384, MEDGE, 384, 771, 1);  // msg_l1 (fused gather)
    gemm(hid, nullptr, F(p + 7), F(p + 6), msgs, nullptr, 0, 1, 384, 384, MEDGE, 384, 384, 0); // msg_l2
    k_agg<<<dim3(NND, GN), 384, 0, stream>>>(msgs, rowstart, order, nb[1]);                    // agg
    gemm(nb[0], nb[1], F(p + 1), F(p + 0), nb[2], nullptr, 1, 0, 384, 384, MNODE, 384, 768, 2); // gate
    gemm(nb[0], nb[1], F(p + 9), F(p + 8), nb[3], nullptr, 1, 0, 384, 384, MNODE, 384, 768, 1); // upd_l1
    gemm(nb[3], nullptr, F(p + 11), F(p + 10), nb[4], nullptr, 0, 0, 384, 384, MNODE, 384, 384, 0); // upd_l2
    k_ln<<<MNODE, 128, 0, stream>>>(nb[2], nb[4], nb[0], F(p + 3), F(p + 2), nb[0], 1);
  }

  // ---- spatial masked attention (sa leaves: 120 k_b..129 v_w) ----
  gemm(nb[0], nullptr, F(127), F(126), nb[1], nullptr, 0, 0, 384, 384, MNODE, 384, 384, 0); // Q
  gemm(nb[0], nullptr, F(121), F(120), nb[2], nullptr, 0, 0, 384, 384, MNODE, 384, 384, 0); // K
  gemm(nb[0], nullptr, F(129), F(128), nb[3], nullptr, 0, 0, 384, 384, MNODE, 384, 384, 0); // V
  k_flash<48, 64><<<dim3(64, 8, 6), 32, 0, stream>>>(nb[1], nb[2], nb[3], nb[4], sscale, maskw);
  gemm(nb[4], nullptr, F(125), F(124), nb[5], nullptr, 0, 0, 384, 384, MNODE, 384, 384, 0); // O-proj
  k_ln<<<MNODE, 128, 0, stream>>>(nb[0], nb[5], nullptr, F(123), F(122), nb[0], 0);

  // ---- temporal attention (ta leaves: 130 k_b..139 v_w) ----
  k_permute_ht<<<(MNODE * HH + 255) / 256, 256, 0, stream>>>(nb[0], nb[1]);  // (t,n)->(n,t)
  gemm(nb[1], nullptr, F(137), F(136), nb[2], nullptr, 0, 0, 384, 384, MNODE, 384, 384, 0); // Q
  gemm(nb[1], nullptr, F(131), F(130), nb[3], nullptr, 0, 0, 384, 384, MNODE, 384, 384, 0); // K
  gemm(nb[1], nullptr, F(139), F(138), nb[4], nullptr, 0, 0, 384, 384, MNODE, 384, 384, 0); // V
  k_tattn<<<(NND * 8 * 6 + 255) / 256, 256, 0, stream>>>(nb[2], nb[3], nb[4], nb[5], sscale);
  gemm(nb[5], nullptr, F(135), F(134), nb[6], nullptr, 0, 0, 384, 384, MNODE, 384, 384, 0); // O-proj
  k_ln<<<MNODE, 128, 0, stream>>>(nb[1], nb[6], nullptr, F(133), F(132), nb[7], 0);

  // ---- head on last timestep (row stride 6*384, offset t=5) ----
  gemm(nb[7] + 5 * 384, nullptr, F(23), F(22), (float*)d_out, nullptr,
       0, 0, 6 * 384, 4, NND, 4, 384, 0);
}
